// StockPredictionModel_41540923687157
// MI455X (gfx1250) — matrix operand order repaired and checked by reading, not compiled, not run on hardware
//
#include <hip/hip_runtime.h>
#include <hip/hip_bf16.h>

typedef __attribute__((ext_vector_type(16))) _Float16 v16h;
typedef __attribute__((ext_vector_type(8)))  float    v8f;

// ---------------------------------------------------------------------------
// helpers
// ---------------------------------------------------------------------------
__device__ inline void atomicMaxFloat(float* addr, float val) {
  if (val >= 0.0f) atomicMax((int*)addr, __float_as_int(val));
  else             atomicMin((unsigned int*)addr, __float_as_uint(val));
}

__device__ inline float sigmoidf(float x) { return 1.0f / (1.0f + __expf(-x)); }

// ---------------------------------------------------------------------------
// WMMA fragment loaders (CDNA5 16-bit layouts, CDNA5 ISA 7.12.2)
// A 16x32 (MxK): lane 0-15 -> M=lane; halves h<8 -> K=g*8+h, h>=8 -> K=16+g*8+(h-8)
//   => per lane: two contiguous 8-float runs at k0+g*8 and k0+g*8+16 -> 4x b128.
// B 32x16 (KxN) from row-major W[Nout,K]: lane col n=lane&15, same K order as A
//   => per lane: two contiguous 8-float runs at k0+g*8 and k0+g*8+16 -> 4x b128.
// NOTE: all GEMMs in this model have K % 32 == 0 (64/128/256/512); no K checks.
// ---------------------------------------------------------------------------
__device__ inline v16h load_a_fast(const float* __restrict__ A, int lda,
                                   int row0, int k0, int la, int g) {
  const float* p = A + (long)(row0 + la) * lda + k0 + g * 8;
  float4 x0 = *(const float4*)(p);
  float4 x1 = *(const float4*)(p + 4);
  float4 x2 = *(const float4*)(p + 16);
  float4 x3 = *(const float4*)(p + 20);
  v16h a;
  a[0]  = (_Float16)x0.x; a[1]  = (_Float16)x0.y; a[2]  = (_Float16)x0.z; a[3]  = (_Float16)x0.w;
  a[4]  = (_Float16)x1.x; a[5]  = (_Float16)x1.y; a[6]  = (_Float16)x1.z; a[7]  = (_Float16)x1.w;
  a[8]  = (_Float16)x2.x; a[9]  = (_Float16)x2.y; a[10] = (_Float16)x2.z; a[11] = (_Float16)x2.w;
  a[12] = (_Float16)x3.x; a[13] = (_Float16)x3.y; a[14] = (_Float16)x3.z; a[15] = (_Float16)x3.w;
  return a;
}

__device__ inline v16h load_b_fast(const float* __restrict__ W, int ldw,
                                   int col0, int k0, int la, int g) {
  const float* p = W + (long)(col0 + la) * ldw + k0 + g * 8;
  float4 x0 = *(const float4*)(p);
  float4 x1 = *(const float4*)(p + 4);
  float4 x2 = *(const float4*)(p + 16);
  float4 x3 = *(const float4*)(p + 20);
  v16h b;
  b[0]  = (_Float16)x0.x; b[1]  = (_Float16)x0.y; b[2]  = (_Float16)x0.z; b[3]  = (_Float16)x0.w;
  b[4]  = (_Float16)x1.x; b[5]  = (_Float16)x1.y; b[6]  = (_Float16)x1.z; b[7]  = (_Float16)x1.w;
  b[8]  = (_Float16)x2.x; b[9]  = (_Float16)x2.y; b[10] = (_Float16)x2.z; b[11] = (_Float16)x2.w;
  b[12] = (_Float16)x3.x; b[13] = (_Float16)x3.y; b[14] = (_Float16)x3.z; b[15] = (_Float16)x3.w;
  return b;
}

// branchless tail variants: clamp index to a valid address, select-zero after.
__device__ inline v16h load_a_gen(const float* __restrict__ A, int lda,
                                  int row0, int k0, int la, int g, int M) {
  int r = row0 + la;
  bool rok = (r < M);
  const float* Ar = A + (long)(rok ? r : (M - 1)) * lda;
#pragma unroll
  for (int z = 0; z < 1; ++z) {}
  v16h a;
#pragma unroll
  for (int j = 0; j < 8; ++j) {
    int kb = k0 + g * 8 + ((j < 4) ? (2 * j) : (16 + 2 * (j - 4)));
    float x0 = Ar[kb];
    float x1 = Ar[kb + 1];
    a[2 * j]     = (_Float16)(rok ? x0 : 0.0f);
    a[2 * j + 1] = (_Float16)(rok ? x1 : 0.0f);
  }
  return a;
}

__device__ inline v16h load_b_gen(const float* __restrict__ W, int ldw,
                                  int col0, int k0, int la, int g, int Nout) {
  int n = col0 + la;
  bool nok = (n < Nout);
  const float* Wr = W + (long)(nok ? n : (Nout - 1)) * ldw;
  v16h b;
#pragma unroll
  for (int h = 0; h < 16; ++h) {
    float x = Wr[k0 + g * 8 + ((h < 8) ? h : (h + 8))];
    b[h] = (_Float16)(nok ? x : 0.0f);
  }
  return b;
}

// ---------------------------------------------------------------------------
// Generic WMMA GEMM: C[M,Nout] = act(A[M,K] @ W[Nout,K]^T + bias)
// block = 256 threads = 8 waves; wave tile 16(M) x 32(N) (two 16x16 WMMA tiles
// sharing the A fragment); block tile 32 x 128.  Requires K % 32 == 0.
// ---------------------------------------------------------------------------
__global__ void k_gemm_wmma(const float* __restrict__ A, const float* __restrict__ W,
                            const float* __restrict__ bias, float* __restrict__ C,
                            int M, int K, int Nout, int act) {
  int wave = threadIdx.x >> 5;
  int lane = threadIdx.x & 31;
  int la   = lane & 15;
  int g    = lane >> 4;
  int row0 = blockIdx.x * 32 + (wave & 1) * 16;
  int colA = blockIdx.y * 128 + (wave >> 1) * 32;
  if (row0 >= M || colA >= Nout) return;          // wave-uniform

  bool rowFull = (row0 + 16 <= M);
  bool has1    = (colA + 16) < Nout;
  bool full1   = (colA + 32 <= Nout);

  v8f acc0 = {}, acc1 = {};

  if (rowFull && full1) {
    // fully in-range: pure b128 loads, 2 WMMA per 32-K slice
    for (int k0 = 0; k0 < K; k0 += 32) {
      v16h a  = load_a_fast(A, K, row0, k0, la, g);
      v16h b0 = load_b_fast(W, K, colA, k0, la, g);
      v16h b1 = load_b_fast(W, K, colA + 16, k0, la, g);
      acc0 = __builtin_amdgcn_wmma_f32_16x16x32_f16(false, a, false, b0,
                                                    (short)0, acc0, false, false);
      acc1 = __builtin_amdgcn_wmma_f32_16x16x32_f16(false, a, false, b1,
                                                    (short)0, acc1, false, false);
    }
  } else {
    for (int k0 = 0; k0 < K; k0 += 32) {
      v16h a  = load_a_gen(A, K, row0, k0, la, g, M);
      v16h b0 = load_b_gen(W, K, colA, k0, la, g, Nout);
      acc0 = __builtin_amdgcn_wmma_f32_16x16x32_f16(false, a, false, b0,
                                                    (short)0, acc0, false, false);
      if (has1) {
        v16h b1 = load_b_gen(W, K, colA + 16, k0, la, g, Nout);
        acc1 = __builtin_amdgcn_wmma_f32_16x16x32_f16(false, a, false, b1,
                                                      (short)0, acc1, false, false);
      }
    }
  }

  // store: C/D layout -> VGPR r holds M = row0 + r + g*8, N = lane&15 (+tile)
  int n0 = colA + la;
  if (n0 < Nout) {
    float bv = bias ? bias[n0] : 0.0f;
#pragma unroll
    for (int r = 0; r < 8; ++r) {
      int m = row0 + r + g * 8;
      if (m < M) {
        float v = acc0[r] + bv;
        if (act) v = fmaxf(v, 0.0f);
        C[(long)m * Nout + n0] = v;
      }
    }
  }
  int n1 = colA + 16 + la;
  if (has1 && n1 < Nout) {
    float bv = bias ? bias[n1] : 0.0f;
#pragma unroll
    for (int r = 0; r < 8; ++r) {
      int m = row0 + r + g * 8;
      if (m < M) {
        float v = acc1[r] + bv;
        if (act) v = fmaxf(v, 0.0f);
        C[(long)m * Nout + n1] = v;
      }
    }
  }
}

// ---------------------------------------------------------------------------
// small utility kernels
// ---------------------------------------------------------------------------
__global__ void k_fill(float* __restrict__ p, float v, long n) {
  long i = blockIdx.x * (long)blockDim.x + threadIdx.x;
  if (i < n) p[i] = v;
}

__global__ void k_reduce_sum(const float* __restrict__ p, long n, float* __restrict__ out) {
  long i = blockIdx.x * (long)blockDim.x + threadIdx.x;
  if (i < n) atomicAdd(out, p[i]);
}

__global__ void k_scale1(float* p, float s) {
  if (blockIdx.x == 0 && threadIdx.x == 0) p[0] *= s;
}

__global__ void k_stats_init(float* stats) {
  int i = threadIdx.x;
  if (i < 5) stats[i] = -1e30f;
  else if (i < 10) stats[i] = 0.0f;
}

// ---------------------------------------------------------------------------
// GRU step kernels.  softmax over nodes (axis=1) per feature (F=5).
// ---------------------------------------------------------------------------
__global__ void k_gru_logits(const float* __restrict__ h, const float* __restrict__ attn,
                             float* __restrict__ l, float* __restrict__ stats, int N) {
  int idx = blockIdx.x * blockDim.x + threadIdx.x;
  if (idx >= N * 5) return;
  int n = idx / 5, f = idx % 5;
  const float* hr = h + (long)n * 128;
  const float* ar = attn + f * 128;
  float s = 0.0f;
  for (int j = 0; j < 128; ++j) s += hr[j] * ar[j];
  l[idx] = s;
  atomicMaxFloat(&stats[f], s);
}

__global__ void k_gru_expsum(float* __restrict__ l, float* __restrict__ stats, int N) {
  int idx = blockIdx.x * blockDim.x + threadIdx.x;
  if (idx >= N * 5) return;
  int f = idx % 5;
  float e = __expf(l[idx] - stats[f]);
  l[idx] = e;
  atomicAdd(&stats[5 + f], e);
}

__global__ void k_gru_update(float* __restrict__ h, const float* __restrict__ gates,
                             const float* __restrict__ xts, const float* __restrict__ w_ih,
                             const float* __restrict__ ex, const float* __restrict__ stats,
                             int N, int t) {
  int idx = blockIdx.x * blockDim.x + threadIdx.x;
  if (idx >= N * 128) return;
  int n = idx >> 7, i = idx & 127;
  const float* xrow  = xts + ((long)n * 32 + t) * 5;
  const float* exrow = ex + (long)n * 5;
  float xg[5];
#pragma unroll
  for (int f = 0; f < 5; ++f) xg[f] = xrow[f] * exrow[f] / stats[5 + f];
  float gr = gates[(long)n * 256 + i];
  float gu = gates[(long)n * 256 + 128 + i];
#pragma unroll
  for (int f = 0; f < 5; ++f) {
    gr += xg[f] * w_ih[i * 5 + f];
    gu += xg[f] * w_ih[(128 + i) * 5 + f];
  }
  float r  = sigmoidf(gr);
  float u  = sigmoidf(gu);
  float hv = h[idx];
  float hh = tanhf(r * hv);
  h[idx] = u * hv + (1.0f - u) * hh;
}

// ---------------------------------------------------------------------------
// GAT kernels.  Edges = E real + N self loops (self-loop weight = mean(ew)).
// ---------------------------------------------------------------------------
__global__ void k_gat_ce(const float* __restrict__ lin_e, const float* __restrict__ a_edge,
                         int heads, int out, float* __restrict__ ce) {
  int hd = threadIdx.x;
  if (hd >= heads) return;
  float s = 0.0f;
  for (int o = 0; o < out; ++o) s += lin_e[hd * out + o] * a_edge[hd * out + o];
  ce[hd] = s;
}

__global__ void k_gat_node_al(const float* __restrict__ hfeat,
                              const float* __restrict__ a_src, const float* __restrict__ a_dst,
                              float* __restrict__ als, float* __restrict__ ald,
                              int N, int heads, int out) {
  int idx = blockIdx.x * blockDim.x + threadIdx.x;
  if (idx >= N * heads) return;
  int n = idx / heads, hd = idx % heads;
  const float* hr = hfeat + (long)n * heads * out + hd * out;
  float ss = 0.0f, sd = 0.0f;
  for (int o = 0; o < out; ++o) {
    float hv = hr[o];
    ss += hv * a_src[hd * out + o];
    sd += hv * a_dst[hd * out + o];
  }
  als[idx] = ss;
  ald[idx] = sd;
}

__global__ void k_gat_edge_alpha(const int* __restrict__ src, const int* __restrict__ dst,
                                 const float* __restrict__ ew, const float* __restrict__ emean,
                                 const float* __restrict__ als, const float* __restrict__ ald,
                                 const float* __restrict__ ce,
                                 float* __restrict__ alpha, float* __restrict__ amax,
                                 int E, int N, int heads) {
  int e = blockIdx.x * blockDim.x + threadIdx.x;
  if (e >= E + N) return;
  int s = (e < E) ? src[e] : (e - E);
  int d = (e < E) ? dst[e] : (e - E);
  float w = (e < E) ? ew[e] : emean[0];
  for (int hd = 0; hd < heads; ++hd) {
    float a = als[s * heads + hd] + ald[d * heads + hd] + w * ce[hd];
    a = (a > 0.0f) ? a : 0.2f * a;                 // leaky_relu 0.2
    alpha[(long)e * heads + hd] = a;
    atomicMaxFloat(&amax[d * heads + hd], a);
  }
}

__global__ void k_gat_edge_exp(const int* __restrict__ dst, float* __restrict__ alpha,
                               const float* __restrict__ amax, float* __restrict__ den,
                               int E, int N, int heads) {
  int e = blockIdx.x * blockDim.x + threadIdx.x;
  if (e >= E + N) return;
  int d = (e < E) ? dst[e] : (e - E);
  for (int hd = 0; hd < heads; ++hd) {
    float ex = __expf(alpha[(long)e * heads + hd] - amax[d * heads + hd]);
    alpha[(long)e * heads + hd] = ex;
    atomicAdd(&den[d * heads + hd], ex);
  }
}

__global__ void k_gat_edge_msg(const int* __restrict__ src, const int* __restrict__ dst,
                               const float* __restrict__ ex, const float* __restrict__ den,
                               const float* __restrict__ hfeat, float* __restrict__ msg,
                               int E, int N, int heads, int out) {
  int  o4    = (out + 3) >> 2;
  long idx   = blockIdx.x * (long)blockDim.x + threadIdx.x;
  long total = (long)(E + N) * heads * o4;
  if (idx >= total) return;
  int  c  = (int)(idx % o4);
  long r  = idx / o4;
  int  hd = (int)(r % heads);
  long e  = r / heads;
  int  s  = (e < E) ? src[e] : (int)(e - E);
  int  d  = (e < E) ? dst[e] : (int)(e - E);
  float w = ex[e * heads + hd] / (den[d * heads + hd] + 1e-16f);
  const float* hs = hfeat + (long)s * heads * out + hd * out;
  float*       md = msg   + (long)d * heads * out + hd * out;
  int ob = c * 4;
  int oe = (ob + 4 < out) ? (ob + 4) : out;
  for (int o = ob; o < oe; ++o) atomicAdd(&md[o], w * hs[o]);
}

__global__ void k_gat_finalize(const float* __restrict__ msg, const float* __restrict__ bias,
                               float* __restrict__ res, int N, int heads, int out,
                               int concat, int act) {
  int C = concat ? heads * out : out;
  long idx = blockIdx.x * (long)blockDim.x + threadIdx.x;
  if (idx >= (long)N * C) return;
  int n = (int)(idx / C), c = (int)(idx % C);
  float v;
  if (concat) v = msg[(long)n * heads * out + c];
  else {
    v = 0.0f;
    for (int hd = 0; hd < heads; ++hd) v += msg[(long)n * heads * out + hd * out + c];
    v /= (float)heads;
  }
  v += bias[c];
  if (act) v = fmaxf(v, 0.0f);
  res[idx] = v;
}

// ---------------------------------------------------------------------------
// cross-attention: one wave per node, 32 market keys, H=128.
// ---------------------------------------------------------------------------
__global__ void k_cross_attn(const float* __restrict__ q, const float* __restrict__ kb,
                             const float* __restrict__ vb, float* __restrict__ outp, int N) {
  int wave = threadIdx.x >> 5;
  int lane = threadIdx.x & 31;
  int n = blockIdx.x * 8 + wave;
  if (n >= N) return;                         // wave-uniform
  const float* qr = q  + (long)n * 128;
  const float* kr = kb + lane * 128;
  float s = 0.0f;
  for (int c = 0; c < 128; ++c) s += qr[c] * kr[c];
  s *= 0.088388347648318447f;                 // 128^-0.5
  float m = s;
  for (int off = 16; off; off >>= 1) m = fmaxf(m, __shfl_xor(m, off, 32));
  float p = __expf(s - m);
  float sum = p;
  for (int off = 16; off; off >>= 1) sum += __shfl_xor(sum, off, 32);
  p /= sum;
  float* orow = outp + (long)n * 128;
  for (int c0 = 0; c0 < 128; c0 += 32) {
    int c = c0 + lane;
    float acc = 0.0f;
    for (int j = 0; j < 32; ++j) acc += __shfl(p, j, 32) * vb[j * 128 + c];
    orow[c] = acc;
  }
}

__global__ void k_concat4(const float* __restrict__ a, const float* __restrict__ b,
                          const float* __restrict__ c, const float* __restrict__ d,
                          float* __restrict__ out, int N) {
  long idx = blockIdx.x * (long)blockDim.x + threadIdx.x;
  if (idx >= (long)N * 512) return;
  int n = (int)(idx >> 9), ch = (int)(idx & 511);
  float v;
  if      (ch < 128) v = a[(long)n * 128 + ch];
  else if (ch < 256) v = b[(long)n * 128 + ch - 128];
  else if (ch < 384) v = c[(long)n * 128 + ch - 256];
  else               v = d[(long)n * 128 + ch - 384];
  out[idx] = v;
}

// ---------------------------------------------------------------------------
// host orchestration
// ---------------------------------------------------------------------------
extern "C" void kernel_launch(void* const* d_in, const int* in_sizes, int n_in,
                              void* d_out, int out_size, void* d_ws, size_t ws_size,
                              hipStream_t stream) {
  (void)in_sizes; (void)n_in; (void)out_size; (void)ws_size;
  const int N = 10000, T = 32, E = 320000;

  const float* x_ts    = (const float*)d_in[0];
  const float* x_graph = (const float*)d_in[1];
  const float* ew      = (const float*)d_in[2];
  const int*   eidx    = (const int*)  d_in[3];
  const int*   src     = eidx;
  const int*   dst     = eidx + E;
  const float* gru_w_ih = (const float*)d_in[4];
  const float* gru_w_hh = (const float*)d_in[5];
  const float* gru_attn = (const float*)d_in[6];
  const float* g1a_W  = (const float*)d_in[7];
  const float* g1a_as = (const float*)d_in[8];
  const float* g1a_ad = (const float*)d_in[9];
  const float* g1a_ae = (const float*)d_in[10];
  const float* g1a_le = (const float*)d_in[11];
  const float* g1a_b  = (const float*)d_in[12];
  const float* g1b_W  = (const float*)d_in[13];
  const float* g1b_as = (const float*)d_in[14];
  const float* g1b_ad = (const float*)d_in[15];
  const float* g1b_ae = (const float*)d_in[16];
  const float* g1b_le = (const float*)d_in[17];
  const float* g1b_b  = (const float*)d_in[18];
  const float* market1 = (const float*)d_in[19];
  const float* market2 = (const float*)d_in[20];
  const float* ca_Wq = (const float*)d_in[21];
  const float* ca_Wk = (const float*)d_in[22];
  const float* ca_Wv = (const float*)d_in[23];
  const float* ca_bq = (const float*)d_in[24];
  const float* ca_bk = (const float*)d_in[25];
  const float* ca_bv = (const float*)d_in[26];
  const float* sa_Wv = (const float*)d_in[29];
  const float* sa_bv = (const float*)d_in[32];
  const float* f1_W  = (const float*)d_in[33];
  const float* f1_as = (const float*)d_in[34];
  const float* f1_ad = (const float*)d_in[35];
  const float* f1_ae = (const float*)d_in[36];
  const float* f1_le = (const float*)d_in[37];
  const float* f1_b  = (const float*)d_in[38];
  const float* f2_W  = (const float*)d_in[39];
  const float* f2_as = (const float*)d_in[40];
  const float* f2_ad = (const float*)d_in[41];
  const float* f2_ae = (const float*)d_in[42];
  const float* f2_le = (const float*)d_in[43];
  const float* f2_b  = (const float*)d_in[44];

  // workspace arena (floats); every region size is a multiple of 4 floats so
  // all GEMM operands stay 16-byte aligned for b128 loads.
  float* W = (float*)d_ws;
  size_t off = 0;
  auto arena = [&](size_t n) { size_t r = off; off += n; return W + r; };
  float* h1    = arena((size_t)N * 128);
  float* xgat  = arena((size_t)N * 128);
  float* rep1  = arena((size_t)N * 128);
  float* rep2  = arena((size_t)N * 128);
  float* catb  = arena((size_t)N * 512);
  float* attb  = arena((size_t)N * 512);
  float* scrA  = arena((size_t)N * 256);       // hfeat / gru gates
  float* scrB  = arena((size_t)N * 256);       // msg / g / y1
  float* alpha = arena((size_t)(E + N) * 2);
  float* als   = arena((size_t)N * 2);
  float* ald   = arena((size_t)N * 2);
  float* amax  = arena((size_t)N * 2);
  float* den   = arena((size_t)N * 2);
  float* gl    = arena((size_t)N * 8);
  float* stats = arena(16);
  float* ce    = arena(4);
  float* emean = arena(4);
  float* kbuf  = arena(32 * 128);
  float* vbuf  = arena(32 * 128);
  float* qbuf  = arena((size_t)N * 128);

  auto fill = [&](float* p, float v, long n) {
    k_fill<<<dim3((unsigned)((n + 255) / 256)), dim3(256), 0, stream>>>(p, v, n);
  };
  auto gemm = [&](const float* Ap, const float* Wp, const float* bp, float* Cp,
                  int M, int Kd, int Nout, int act) {
    dim3 grid((unsigned)((M + 31) / 32), (unsigned)((Nout + 127) / 128));
    k_gemm_wmma<<<grid, dim3(256), 0, stream>>>(Ap, Wp, bp, Cp, M, Kd, Nout, act);
  };
  auto gat = [&](const float* x, int inF, int heads, int out,
                 const float* Wm, const float* asr, const float* ads, const float* aed,
                 const float* lin, const float* bias, int concat, int act,
                 float* hfeat, float* msg, float* res) {
    gemm(x, Wm, nullptr, hfeat, N, inF, heads * out, 0);
    k_gat_ce<<<1, 32, 0, stream>>>(lin, aed, heads, out, ce);
    int NH = N * heads;
    k_gat_node_al<<<dim3((NH + 255) / 256), dim3(256), 0, stream>>>(
        hfeat, asr, ads, als, ald, N, heads, out);
    fill(amax, -1e30f, NH);
    fill(den, 0.0f, NH);
    fill(msg, 0.0f, (long)N * heads * out);
    int EN = E + N;
    k_gat_edge_alpha<<<dim3((EN + 255) / 256), dim3(256), 0, stream>>>(
        src, dst, ew, emean, als, ald, ce, alpha, amax, E, N, heads);
    k_gat_edge_exp<<<dim3((EN + 255) / 256), dim3(256), 0, stream>>>(
        dst, alpha, amax, den, E, N, heads);
    long tot = (long)EN * heads * ((out + 3) / 4);
    k_gat_edge_msg<<<dim3((unsigned)((tot + 255) / 256)), dim3(256), 0, stream>>>(
        src, dst, alpha, den, hfeat, msg, E, N, heads, out);
    long nc = (long)N * (concat ? heads * out : out);
    k_gat_finalize<<<dim3((unsigned)((nc + 255) / 256)), dim3(256), 0, stream>>>(
        msg, bias, res, N, heads, out, concat, act);
  };

  // ---- edge-attr mean (scalar) ----
  fill(emean, 0.0f, 1);
  k_reduce_sum<<<dim3((E + 255) / 256), dim3(256), 0, stream>>>(ew, E, emean);
  k_scale1<<<1, 32, 0, stream>>>(emean, 1.0f / (float)E);

  // ---- GRU scan (32 sequential steps) ----
  fill(h1, 0.0f, (long)N * 128);
  for (int t = 0; t < T; ++t) {
    k_stats_init<<<1, 32, 0, stream>>>(stats);
    k_gru_logits<<<dim3((N * 5 + 255) / 256), dim3(256), 0, stream>>>(h1, gru_attn, gl, stats, N);
    k_gru_expsum<<<dim3((N * 5 + 255) / 256), dim3(256), 0, stream>>>(gl, stats, N);
    gemm(h1, gru_w_hh, nullptr, scrA, N, 128, 256, 0);      // h @ W_hh^T  (WMMA)
    k_gru_update<<<dim3((N * 128 + 255) / 256), dim3(256), 0, stream>>>(
        h1, scrA, x_ts, gru_w_ih, gl, stats, N, t);
  }

  // ---- GAT stack: g = relu(gat1a(x_graph)); x_gat = gat1b(g) ----
  gat(x_graph, 64, 2, 128, g1a_W, g1a_as, g1a_ad, g1a_ae, g1a_le, g1a_b, 1, 1, scrA, scrB, scrB);
  gat(scrB, 256, 1, 128, g1b_W, g1b_as, g1b_ad, g1b_ae, g1b_le, g1b_b, 0, 0, scrA, xgat, xgat);

  // ---- cross attention vs market1 (rep1) and market2 (rep2) ----
  gemm(market1, ca_Wk, ca_bk, kbuf, 32, 128, 128, 0);
  gemm(market1, ca_Wv, ca_bv, vbuf, 32, 128, 128, 0);
  gemm(h1, ca_Wq, ca_bq, qbuf, N, 128, 128, 0);
  k_cross_attn<<<dim3((N + 7) / 8), dim3(256), 0, stream>>>(qbuf, kbuf, vbuf, rep1, N);

  gemm(market2, ca_Wk, ca_bk, kbuf, 32, 128, 128, 0);
  gemm(market2, ca_Wv, ca_bv, vbuf, 32, 128, 128, 0);
  gemm(xgat, ca_Wq, ca_bq, qbuf, N, 128, 128, 0);
  k_cross_attn<<<dim3((N + 7) / 8), dim3(256), 0, stream>>>(qbuf, kbuf, vbuf, rep2, N);

  // ---- cat = [h1, x_gat, rep1, rep2]; self-attn over 1 token == V projection ----
  k_concat4<<<dim3((unsigned)(((long)N * 512 + 255) / 256)), dim3(256), 0, stream>>>(
      h1, xgat, rep1, rep2, catb, N);
  gemm(catb, sa_Wv, sa_bv, attb, N, 512, 512, 0);

  // ---- final GATs: y1 = relu(f1(att)); out = relu(f2(y1))[:,0] ----
  gat(attb, 512, 1, 64, f1_W, f1_as, f1_ad, f1_ae, f1_le, f1_b, 1, 1, scrA, scrB, scrB);
  gat(scrB, 64, 1, 1, f2_W, f2_as, f2_ad, f2_ae, f2_le, f2_b, 0, 1, scrA, rep1, (float*)d_out);
}
